// MultiheadAttention_17102559772674
// MI455X (gfx1250) — matrix operand order repaired and checked by reading, not compiled, not run on hardware
//
#include <hip/hip_runtime.h>
#include <math.h>

#define Tdim 1024
#define Bdim 4
#define Sdim 1024
#define Edim 1024
#define Hn   16
#define HD   64
#define BH   (Bdim * Hn)

typedef __attribute__((ext_vector_type(16))) _Float16 v16h;
typedef __attribute__((ext_vector_type(8)))  _Float16 v8h;
typedef __attribute__((ext_vector_type(8)))  float    v8f;

#define WMMA_F32_F16(a, b, c) \
  __builtin_amdgcn_wmma_f32_16x16x32_f16(false, (a), false, (b), (short)0, (c), false, false)

// ---- CDNA5 WMMA fragment loaders (wave32 layouts per ISA 7.12.2) ----
// A (16x32 f16, MxK): lanes 0-15 row m hold K=0..7 (v0-3) and K=16..23 (v4-7);
//                     lanes 16-31 row m hold K=8..15 and K=24..31.
__device__ __forceinline__ v16h load_a_frag(const _Float16* A, int lda, int lane) {
  const int m = lane & 15;
  const int k = (lane & 16) ? 8 : 0;
  v8h c0 = *(const v8h*)(A + (size_t)m * lda + k);
  v8h c1 = *(const v8h*)(A + (size_t)m * lda + k + 16);
  v16h r;
#pragma unroll
  for (int i = 0; i < 8; ++i) { r[i] = c0[i]; r[i + 8] = c1[i]; }
  return r;
}

// B (32x16 f16, KxN) supplied as Bt[N][K] (column n of B == row n of Bt):
// lanes 0-15 hold column n, K=0..7 and K=16..23; lanes 16-31 hold K=8..15 and K=24..31.
__device__ __forceinline__ v16h load_b_frag(const _Float16* Bt, int ldb, int lane) {
  const int n = lane & 15;
  const int k = (lane & 16) ? 8 : 0;
  v8h c0 = *(const v8h*)(Bt + (size_t)n * ldb + k);
  v8h c1 = *(const v8h*)(Bt + (size_t)n * ldb + k + 16);
  v16h r;
#pragma unroll
  for (int i = 0; i < 8; ++i) { r[i] = c0[i]; r[i + 8] = c1[i]; }
  return r;
}

// ---- 64x32 register-blocked GEMM core: acc += A[64,K] * Bt[32,K]^T ----
// 8 WMMAs per K-step from 12 b128 loads (1.5 loads/WMMA); ~170 VGPRs, spill-free.
// Prefetch is unconditional: CDNA5 prefetch is speculative (bad translations are
// dropped, ISA 10.5), so running past the tile end is harmless and keeps the
// K-loop a single basic block (no branch / hazard-nop overhead per iteration).
__device__ __forceinline__ void gemm64x32(const _Float16* __restrict__ A, int lda,
                                          const _Float16* __restrict__ Bt, int ldb,
                                          int K, int lane, v8f acc[4][2]) {
  const int pl = lane & 15;  // lane's resident row for prefetch
  for (int k0 = 0; k0 < K; k0 += 32) {
    __builtin_prefetch(A + (size_t)pl * lda + k0 + 512, 0, 1);   // global_prefetch_b8
    __builtin_prefetch(Bt + (size_t)pl * ldb + k0 + 512, 0, 1);  // ~16 K-tiles ahead
    v16h a[4], b[2];
#pragma unroll
    for (int i = 0; i < 4; ++i)
      a[i] = load_a_frag(A + (size_t)(i * 16) * lda + k0, lda, lane);
#pragma unroll
    for (int j = 0; j < 2; ++j)
      b[j] = load_b_frag(Bt + (size_t)(j * 16) * ldb + k0, ldb, lane);
#pragma unroll
    for (int i = 0; i < 4; ++i)
#pragma unroll
      for (int j = 0; j < 2; ++j)
        acc[i][j] = WMMA_F32_F16(a[i], b[j], acc[i][j]);
  }
}

// ---- elementwise f32 -> f16 conversion ----
__global__ void cvt_f32_to_f16(const float* __restrict__ src, _Float16* __restrict__ dst, int n) {
  int i = blockIdx.x * blockDim.x + threadIdx.x;
  if (i < n) dst[i] = (_Float16)src[i];
}

// ---- QKV projection: X[M,E] @ W[z]^T + b -> q16 / k16 / vT16 (f16) ----
__global__ void __launch_bounds__(32, 1)
qkv_proj_kernel(const _Float16* __restrict__ xq,
                const _Float16* __restrict__ xk,
                const _Float16* __restrict__ xv,
                const _Float16* __restrict__ w16,   // [3E, E]
                const float*    __restrict__ bias,  // [3E]
                _Float16* __restrict__ q16,         // [BH, T, HD], pre-scaled
                _Float16* __restrict__ k16,         // [BH, S, HD]
                _Float16* __restrict__ vT16) {      // [BH, HD, S]
  const int lane = threadIdx.x;
  const int m0 = blockIdx.x * 64;          // rows (t*B + b)
  const int n0 = blockIdx.y * 32;          // output feature tile
  const int z  = blockIdx.z;               // 0=q, 1=k, 2=v
  const _Float16* X  = (z == 0) ? xq : (z == 1) ? xk : xv;
  const _Float16* A  = X + (size_t)m0 * Edim;
  const _Float16* Bt = w16 + (size_t)z * Edim * Edim + (size_t)n0 * Edim;

  v8f acc[4][2] = {};
  gemm64x32(A, Edim, Bt, Edim, Edim, lane, acc);

  const int n = lane & 15;
  const int rbase = (lane >> 4) * 8;       // C/D layout: lanes 16-31 hold M=8..15
#pragma unroll
  for (int j = 0; j < 2; ++j) {
    const int e  = n0 + j * 16 + n;
    const int h  = e >> 6;
    const int hd = e & 63;
    const float bn = bias[z * Edim + e];
#pragma unroll
    for (int i = 0; i < 4; ++i) {
#pragma unroll
      for (int jj = 0; jj < 8; ++jj) {
        const int m = m0 + i * 16 + rbase + jj;  // m = t*B + b  (B == 4)
        const int t = m >> 2;
        const int b = m & 3;
        const int bh = b * Hn + h;
        float v = acc[i][j][jj] + bn;
        if (z == 0) {
          q16[((size_t)bh * Tdim + t) * HD + hd] = (_Float16)(v * 0.125f);  // 1/sqrt(64)
        } else if (z == 1) {
          k16[((size_t)bh * Sdim + t) * HD + hd] = (_Float16)v;
        } else {
          vT16[((size_t)bh * HD + hd) * Sdim + t] = (_Float16)v;            // transposed
        }
      }
    }
  }
}

// ---- scores = Q K^T + mask, padding -> -inf, softmax -> p16 (f16) ----
// One wave handles 32 query rows x full S; Q fragments stay in registers.
__global__ void __launch_bounds__(32, 1)
scores_softmax_kernel(const _Float16* __restrict__ q16,
                      const _Float16* __restrict__ k16,
                      const float* __restrict__ attn_mask, // [T,S]
                      const int*   __restrict__ kpm,       // [B,S]
                      _Float16* __restrict__ p16) {        // [BH,T,S]
  extern __shared__ float smem[];          // [32][Sdim] = 128 KB (<= 320 KB WGP LDS)
  const int lane = threadIdx.x;
  const int bh = blockIdx.x;
  const int t0 = blockIdx.y * 32;
  const int b  = bh >> 4;

  const _Float16* Aq = q16 + ((size_t)bh * Tdim + t0) * HD;
  v16h aq[2][2];                           // [row tile][K half of head dim]
#pragma unroll
  for (int i = 0; i < 2; ++i) {
    aq[i][0] = load_a_frag(Aq + (size_t)(i * 16) * HD, HD, lane);
    aq[i][1] = load_a_frag(Aq + (size_t)(i * 16) * HD + 32, HD, lane);
  }

  const int n = lane & 15;
  const int rbase = (lane >> 4) * 8;

  for (int st = 0; st < Sdim / 16; ++st) {
    const _Float16* Bk = k16 + ((size_t)bh * Sdim + st * 16) * HD;
    v16h b0 = load_b_frag(Bk, HD, lane);
    v16h b1 = load_b_frag(Bk + 32, HD, lane);
    v8f c[2] = {};
#pragma unroll
    for (int i = 0; i < 2; ++i) {
      c[i] = WMMA_F32_F16(aq[i][0], b0, c[i]);
      c[i] = WMMA_F32_F16(aq[i][1], b1, c[i]);
    }
    const int s = st * 16 + n;
    const bool padded = (kpm[b * Sdim + s] != 0);
#pragma unroll
    for (int i = 0; i < 2; ++i) {
#pragma unroll
      for (int j = 0; j < 8; ++j) {
        const int r = i * 16 + rbase + j;
        float v = c[i][j] + attn_mask[(size_t)(t0 + r) * Sdim + s];
        smem[r * Sdim + s] = padded ? -INFINITY : v;
      }
    }
  }
  __syncthreads();

  for (int r = 0; r < 32; ++r) {
    float* row = smem + r * Sdim;
    float mx = -INFINITY;
    for (int c = lane; c < Sdim; c += 32) mx = fmaxf(mx, row[c]);
#pragma unroll
    for (int off = 16; off > 0; off >>= 1) mx = fmaxf(mx, __shfl_xor(mx, off, 32));
    float sum = 0.f;
    for (int c = lane; c < Sdim; c += 32) {
      float v = row[c];
      float e = (v == -INFINITY) ? 0.f : __expf(v - mx);  // NaN-safe for fully-masked rows
      row[c] = e;
      sum += e;
    }
#pragma unroll
    for (int off = 16; off > 0; off >>= 1) sum += __shfl_xor(sum, off, 32);
    const float inv = (sum > 0.f) ? (1.f / sum) : 0.f;
    _Float16* prow = p16 + ((size_t)bh * Tdim + t0 + r) * Sdim;
    for (int c = lane; c < Sdim; c += 32) prow[c] = (_Float16)(row[c] * inv);
  }
}

// ---- context = P @ V  -> attn16 [T*B, E] (f16), 64x32 per wave ----
__global__ void __launch_bounds__(32, 1)
attn_v_kernel(const _Float16* __restrict__ p16,    // [BH,T,S]
              const _Float16* __restrict__ vT16,   // [BH,HD,S]
              _Float16* __restrict__ attn16) {     // [T*B, E]
  const int lane = threadIdx.x;
  const int bh = blockIdx.x;
  const int t0 = blockIdx.y * 64;
  const int nb = blockIdx.z * 32;          // head-dim half
  const int b = bh >> 4, h = bh & 15;
  const _Float16* Ap = p16 + ((size_t)bh * Tdim + t0) * Sdim;
  const _Float16* Bv = vT16 + ((size_t)bh * HD + nb) * Sdim;

  v8f acc[4][2] = {};
  gemm64x32(Ap, Sdim, Bv, Sdim, Sdim, lane, acc);

  const int n = lane & 15;
  const int rbase = (lane >> 4) * 8;
#pragma unroll
  for (int i = 0; i < 4; ++i) {
#pragma unroll
    for (int j = 0; j < 2; ++j) {
#pragma unroll
      for (int jj = 0; jj < 8; ++jj) {
        const int t = t0 + i * 16 + rbase + jj;
        const int e = h * HD + nb + j * 16 + n;
        attn16[((size_t)t * Bdim + b) * Edim + e] = (_Float16)acc[i][j][jj];
      }
    }
  }
}

// ---- out = attn @ out_w^T + out_b  (f32 to d_out), 64x32 per wave ----
__global__ void __launch_bounds__(32, 1)
out_proj_kernel(const _Float16* __restrict__ attn16,
                const _Float16* __restrict__ ow16,  // [E,E]
                const float* __restrict__ out_b,
                float* __restrict__ out) {
  const int lane = threadIdx.x;
  const int m0 = blockIdx.x * 64;
  const int n0 = blockIdx.y * 32;
  const _Float16* A  = attn16 + (size_t)m0 * Edim;
  const _Float16* Bt = ow16 + (size_t)n0 * Edim;

  v8f acc[4][2] = {};
  gemm64x32(A, Edim, Bt, Edim, Edim, lane, acc);

  const int n = lane & 15;
  const int rbase = (lane >> 4) * 8;
#pragma unroll
  for (int j = 0; j < 2; ++j) {
    const int e = n0 + j * 16 + n;
    const float bn = out_b[e];
#pragma unroll
    for (int i = 0; i < 4; ++i) {
#pragma unroll
      for (int jj = 0; jj < 8; ++jj) {
        const int m = m0 + i * 16 + rbase + jj;
        out[(size_t)m * Edim + e] = acc[i][j][jj] + bn;
      }
    }
  }
}

// ---- weights_avg = mean over heads of p16 ----
__global__ void weights_avg_kernel(const _Float16* __restrict__ p16,
                                   float* __restrict__ wavg) {      // [B,T,S]
  const size_t idx = (size_t)blockIdx.x * blockDim.x + threadIdx.x;
  const int s = (int)(idx & (Sdim - 1));
  const int t = (int)((idx >> 10) & (Tdim - 1));
  const int b = (int)(idx >> 20);
  float sum = 0.f;
#pragma unroll
  for (int h = 0; h < Hn; ++h)
    sum += (float)p16[(((size_t)(b * Hn + h)) * Tdim + t) * Sdim + s];
  wavg[idx] = sum * (1.f / Hn);
}

extern "C" void kernel_launch(void* const* d_in, const int* in_sizes, int n_in,
                              void* d_out, int out_size, void* d_ws, size_t ws_size,
                              hipStream_t stream) {
  const float* query     = (const float*)d_in[0];
  const float* key_      = (const float*)d_in[1];
  const float* value     = (const float*)d_in[2];
  const float* attn_mask = (const float*)d_in[3];
  const int*   kpm       = (const int*)d_in[4];
  const float* ipw       = (const float*)d_in[5];
  const float* ipb       = (const float*)d_in[6];
  const float* out_w     = (const float*)d_in[7];
  const float* out_b     = (const float*)d_in[8];
  float* out  = (float*)d_out;
  float* wavg = out + (size_t)Tdim * Bdim * Edim;

  char* ws = (char*)d_ws;
  size_t off = 0;
  auto alloc = [&](size_t elems) -> _Float16* {
    _Float16* p = (_Float16*)(ws + off);
    off = (off + elems * sizeof(_Float16) + 255) & ~(size_t)255;
    return p;
  };
  _Float16* xq16   = alloc((size_t)Tdim * Bdim * Edim);
  _Float16* xk16   = alloc((size_t)Sdim * Bdim * Edim);
  _Float16* xv16   = alloc((size_t)Sdim * Bdim * Edim);
  _Float16* w16    = alloc((size_t)3 * Edim * Edim);
  _Float16* ow16   = alloc((size_t)Edim * Edim);
  _Float16* q16    = alloc((size_t)BH * Tdim * HD);
  _Float16* k16    = alloc((size_t)BH * Sdim * HD);
  _Float16* vT16   = alloc((size_t)BH * HD * Sdim);
  _Float16* attn16 = alloc((size_t)Tdim * Bdim * Edim);
  _Float16* p16    = alloc((size_t)BH * Tdim * Sdim);

  auto cvt = [&](const float* src, _Float16* dst, size_t n) {
    cvt_f32_to_f16<<<dim3((unsigned)((n + 255) / 256)), dim3(256), 0, stream>>>(src, dst, (int)n);
  };
  cvt(query, xq16, (size_t)Tdim * Bdim * Edim);
  cvt(key_,  xk16, (size_t)Sdim * Bdim * Edim);
  cvt(value, xv16, (size_t)Sdim * Bdim * Edim);
  cvt(ipw,   w16,  (size_t)3 * Edim * Edim);
  cvt(out_w, ow16, (size_t)Edim * Edim);

  qkv_proj_kernel<<<dim3(Tdim * Bdim / 64, Edim / 32, 3), dim3(32), 0, stream>>>(
      xq16, xk16, xv16, w16, ipb, q16, k16, vT16);

  scores_softmax_kernel<<<dim3(BH, Tdim / 32), dim3(32), 32 * Sdim * sizeof(float), stream>>>(
      q16, k16, attn_mask, kpm, p16);

  attn_v_kernel<<<dim3(BH, Tdim / 64, 2), dim3(32), 0, stream>>>(p16, vT16, attn16);

  out_proj_kernel<<<dim3(Tdim * Bdim / 64, Edim / 32), dim3(32), 0, stream>>>(
      attn16, ow16, out_b, out);

  weights_avg_kernel<<<dim3(Bdim * Tdim * Sdim / 256), dim3(256), 0, stream>>>(p16, wavg);
}
